// LearnableAlignmentFusion_3951369912785
// MI455X (gfx1250) — matrix operand order repaired and checked by reading, not compiled, not run on hardware
//
#include <hip/hip_runtime.h>

// ---------------------------------------------------------------------------
// LearnableAlignmentFusion on MI455X (gfx1250, wave32, WMMA)
// GEMMs (conv1-as-im2col, fusion 1x1, final linear) run on
// v_wmma_f32_16x16x32_bf16. A-tiles (and where possible B-tiles) are staged
// into LDS with GLOBAL_LOAD_ASYNC_TO_LDS_B128 (ASYNCcnt path, no VGPR
// staging). flow conv2 + bilinear warp are VALU kernels.
// ---------------------------------------------------------------------------

typedef __bf16 bf16_t;
typedef __attribute__((ext_vector_type(16))) __bf16 v16bf;
typedef __attribute__((ext_vector_type(8)))  __bf16 v8bfp;
typedef __attribute__((ext_vector_type(8)))  float  v8f;

// problem constants
#define B_    64
#define F_    5
#define C_    256
#define H_    8
#define W_    64
#define PX_   (H_ * W_)          // 512
#define PAIRS (B_ * (F_ - 1))    // 256
#define K1_   (9 * 2 * C_)       // 4608  (im2col K for conv1)
#define KF_   (F_ * C_)          // 1280  (fusion K)
#define KL_   (C_ * H_)          // 2048  (linear K)

// workspace layout (bytes, all 256B aligned)
static constexpr size_t OFF_XBF    = 0;            // 320*256*512 bf16  (83,886,080)
static constexpr size_t OFF_A1     = 83886080;     // 128*4608   bf16  ( 1,179,648)
static constexpr size_t OFF_HID    = 85065728;     // 256*128*512 bf16 (33,554,432)
static constexpr size_t OFF_SAMP   = 118620160;    // 256*512*4  f32   ( 2,097,152)
static constexpr size_t OFF_FULL   = 120717312;    // 64*1280*512 bf16 (83,886,080)
static constexpr size_t OFF_FUSW   = 204603392;    // 256*1280   bf16  (   655,360)
static constexpr size_t OFF_FUSEDT = 205258752;    // 64*64*2048 bf16  (16,777,216)
static constexpr size_t OFF_LINWT  = 222035968;    // 2048*256   bf16  ( 1,048,576)

#define WSP(base, off, T) ((T*)((char*)(base) + (off)))

// LDS tile stride (bf16 elements): 32 K + 8 pad; row pitch 80B = 5x16B so
// b128 async stores stay 16B-aligned while dodging bank conflicts.
#define LDT 40

// ---------------- CDNA5 async global->LDS copy (ASYNCcnt) ------------------
// GV mode: 64-bit global address in VGPR pair, LDS byte offset in VDST VGPR.
// Low 32 bits of a flat LDS pointer are the LDS offset (aperture rule).
__device__ __forceinline__ void async_copy_b128(const bf16_t* lds_dst,
                                                const bf16_t* gsrc) {
  unsigned lds_off = (unsigned)(unsigned long long)lds_dst;
  asm volatile("global_load_async_to_lds_b128 %0, %1, off"
               :: "v"(lds_off), "v"(gsrc)
               : "memory");
}
__device__ __forceinline__ void wait_async0() {
  asm volatile("s_wait_asynccnt 0x0" ::: "memory");
}

// ---------------- WMMA fragment helpers (CDNA5 wave32 layouts) -------------
// A (16x32 bf16, M x K): lane&15 = M row; lane>>4 = g selects K halves:
//   element e<8  -> K = g*8 + e ;  e>=8 -> K = 16 + g*8 + (e-8)
__device__ __forceinline__ v16bf frag_a(const bf16_t* As, int rowBase, int lane) {
  const int r = rowBase + (lane & 15);
  const int g = (lane >> 4) & 1;
  const bf16_t* p = As + r * LDT + g * 8;
  v16bf a;
#pragma unroll
  for (int j = 0; j < 8; ++j) { a[j] = p[j]; a[j + 8] = p[j + 16]; }
  return a;
}
// B (32x16 bf16, K x N) staged N-major in LDS: lane&15 = N col; lane>>4 = g;
//   element e<8 -> K = g*8 + e ;  e>=8 -> K = 16 + g*8 + (e-8)
__device__ __forceinline__ v16bf frag_b(const bf16_t* Bs, int colBase, int lane) {
  const int n = colBase + (lane & 15);
  const int g = (lane >> 4) & 1;
  const bf16_t* p = Bs + n * LDT + g * 8;
  v16bf b;
#pragma unroll
  for (int j = 0; j < 8; ++j) { b[j] = p[j]; b[j + 8] = p[j + 16]; }
  return b;
}
__device__ __forceinline__ v8f wmma_bf16(v16bf a, v16bf b, v8f c) {
  // D = A*B + C ; f32 accumulate
  return __builtin_amdgcn_wmma_f32_16x16x32_bf16(false, a, false, b, (short)0, c,
                                                 false, false);
}
__device__ __forceinline__ v8f vzero8() {
  v8f z = {0.f, 0.f, 0.f, 0.f, 0.f, 0.f, 0.f, 0.f};
  return z;
}

// A tile (128 rows x 32 K) async copy: 512 b128 items, 2 per thread.
__device__ __forceinline__ void stage_a_async(bf16_t* As, const bf16_t* Ag,
                                              long lda, int k0, int tid) {
#pragma unroll
  for (int i = 0; i < 2; ++i) {
    int it = i * 256 + tid;
    int row = it >> 2, oct = it & 3;          // 4 octets of 8 bf16 per row
    async_copy_b128(&As[row * LDT + oct * 8], Ag + (long)row * lda + k0 + oct * 8);
  }
}

// B tile (32 K x 128 N, global row-major in N) -> LDS N-major:
// vectorized b128 global load + 8x ds_store_b16 transpose scatter.
__device__ __forceinline__ void stage_b_vec(bf16_t* Bs, const bf16_t* Bg,
                                            long ldb, int k0, int nBase, int tid) {
#pragma unroll
  for (int i = 0; i < 2; ++i) {
    int it = i * 256 + tid;                    // 512 items: 32 kk x 16 octets
    int kk = it >> 4, oct = it & 15;
    int nn0 = oct * 8;
    v8bfp v = *(const v8bfp*)(Bg + (long)(k0 + kk) * ldb + nBase + nn0);
#pragma unroll
    for (int u = 0; u < 8; ++u) Bs[(nn0 + u) * LDT + kk] = v[u];
  }
}

// ---------------- conversion / layout kernels ------------------------------
__global__ __launch_bounds__(256) void k_cvt_bf16(const float* __restrict__ in,
                                                  bf16_t* __restrict__ out, int n) {
  int i = blockIdx.x * 256 + threadIdx.x;
  if (i < n) out[i] = (bf16_t)in[i];
}

// flow_w1 (128,512,3,3) -> A1[o][tap*512+ci]  (GEMM A, row-major, lda=4608)
__global__ __launch_bounds__(256) void k_cvt_w1(const float* __restrict__ w1,
                                                bf16_t* __restrict__ A1) {
  int i = blockIdx.x * 256 + threadIdx.x;
  if (i >= 128 * K1_) return;
  int o = i / K1_, r = i - o * K1_;
  int tap = r >> 9, ci = r & 511;
  A1[i] = (bf16_t)w1[((size_t)(o * 512 + ci)) * 9 + tap];
}

// lin_w (256,2048) -> lwT[k][co]  (GEMM B, K x N)
__global__ __launch_bounds__(256) void k_cvt_linw(const float* __restrict__ lw,
                                                  bf16_t* __restrict__ lwT) {
  int i = blockIdx.x * 256 + threadIdx.x;
  if (i >= KL_ * C_) return;
  int k = i >> 8, co = i & 255;
  lwT[i] = (bf16_t)lw[(size_t)co * KL_ + k];
}

// center frame -> FULL slot 2
__global__ __launch_bounds__(256) void k_center(const bf16_t* __restrict__ xbf,
                                                bf16_t* __restrict__ full) {
  int i = blockIdx.x * 256 + threadIdx.x;   // 64*256*512
  if (i >= B_ * C_ * PX_) return;
  int b = i >> 17;                          // 256*512 = 1<<17
  int rem = i & 131071;                     // ch*512 + n
  full[(size_t)b * KF_ * PX_ + (size_t)2 * C_ * PX_ + rem] =
      xbf[(size_t)(b * F_ + 2) * C_ * PX_ + rem];
}

// ---------------- conv1: im2col GEMM, M=128 N=512 K=4608 per pair ----------
__global__ __launch_bounds__(256) void k_conv1(const bf16_t* __restrict__ xbf,
                                               const bf16_t* __restrict__ A1,
                                               const float* __restrict__ b1,
                                               bf16_t* __restrict__ hid) {
  __shared__ bf16_t As[128 * LDT];
  __shared__ bf16_t Bs[128 * LDT];
  const int tid = threadIdx.x, lane = tid & 31, wave = tid >> 5;
  const int wm = wave >> 1, wn = wave & 1;
  const int pair = blockIdx.y;
  const int nBase = blockIdx.x * 128;
  const int b = pair >> 2, j = pair & 3;
  const int t = (j < 2) ? j : j + 1;                      // neighbor frame
  const bf16_t* xc = xbf + (size_t)(b * F_ + 2) * C_ * PX_;  // center
  const bf16_t* xn = xbf + (size_t)(b * F_ + t) * C_ * PX_;  // neighbor

  v8f acc[2][4];
#pragma unroll
  for (int i = 0; i < 2; ++i)
#pragma unroll
    for (int jj = 0; jj < 4; ++jj) acc[i][jj] = vzero8();

  for (int k0 = 0; k0 < K1_; k0 += 32) {
    // stage A tile 128x32 via async global->LDS (ASYNCcnt)
    stage_a_async(As, A1, K1_, k0, tid);
    // build im2col B tile 32x128 (staged N-major), zero-padded borders
#pragma unroll
    for (int i = 0; i < 16; ++i) {
      int ei = i * 256 + tid;
      int kk = ei >> 7, nn = ei & 127;
      int k = k0 + kk;
      int tap = k >> 9, ci = k & 511;
      int dy = tap / 3 - 1, dx = tap % 3 - 1;
      int n = nBase + nn;
      int y = (n >> 6) + dy, x = (n & 63) + dx;
      bf16_t v = (bf16_t)0.0f;
      if ((unsigned)y < (unsigned)H_ && (unsigned)x < (unsigned)W_) {
        const bf16_t* src = (ci < C_) ? xc : xn;
        v = src[(size_t)(ci & 255) * PX_ + y * W_ + x];
      }
      Bs[nn * LDT + kk] = v;
    }
    wait_async0();
    __syncthreads();
    v16bf af[2], bf[4];
#pragma unroll
    for (int i = 0; i < 2; ++i) af[i] = frag_a(As, wm * 32 + i * 16, lane);
#pragma unroll
    for (int jj = 0; jj < 4; ++jj) bf[jj] = frag_b(Bs, wn * 64 + jj * 16, lane);
#pragma unroll
    for (int i = 0; i < 2; ++i)
#pragma unroll
      for (int jj = 0; jj < 4; ++jj) acc[i][jj] = wmma_bf16(af[i], bf[jj], acc[i][jj]);
    __syncthreads();
  }
  // epilogue: bias + relu -> hid[pair][m][n] bf16
  bf16_t* outp = hid + (size_t)pair * 128 * PX_;
  const int g = (lane >> 4) & 1, cl = lane & 15;
#pragma unroll
  for (int i = 0; i < 2; ++i)
#pragma unroll
    for (int jj = 0; jj < 4; ++jj) {
      int col = nBase + wn * 64 + jj * 16 + cl;
#pragma unroll
      for (int r = 0; r < 8; ++r) {
        int m = wm * 32 + i * 16 + g * 8 + r;
        float v = acc[i][jj][r] + b1[m];
        outp[(size_t)m * PX_ + col] = (bf16_t)fmaxf(v, 0.0f);
      }
    }
}

// ---------------- conv2 (128->2, 3x3) + sampling params --------------------
__global__ __launch_bounds__(256) void k_flow2(const bf16_t* __restrict__ hid,
                                               const float* __restrict__ w2,
                                               const float* __restrict__ b2,
                                               float* __restrict__ samp) {
  __shared__ float w2s[2 * 128 * 9];
  for (int i = threadIdx.x; i < 2304; i += 256) w2s[i] = w2[i];
  __syncthreads();
  int idx = blockIdx.x * 256 + threadIdx.x;        // 256*512
  int p = idx >> 9, n = idx & 511, y = n >> 6, x = n & 63;
  const bf16_t* hp0 = hid + (size_t)p * 128 * PX_;
  float a0 = b2[0], a1 = b2[1];
  for (int ky = 0; ky < 3; ++ky) {
    int yy = y + ky - 1;
    if ((unsigned)yy >= (unsigned)H_) continue;
    for (int kx = 0; kx < 3; ++kx) {
      int xx = x + kx - 1;
      if ((unsigned)xx >= (unsigned)W_) continue;
      const bf16_t* hp = hp0 + yy * W_ + xx;
      int wb = ky * 3 + kx;
#pragma unroll 8
      for (int ci = 0; ci < 128; ++ci) {
        float hv = (float)hp[(size_t)ci * PX_];
        a0 = fmaf(hv, w2s[ci * 9 + wb], a0);
        a1 = fmaf(hv, w2s[1152 + ci * 9 + wb], a1);
      }
    }
  }
  // grid = base + flow ; ix = x + fx*31.5 (clipped), iy = y + fy*3.5
  float ix = fminf(fmaxf((float)x + a0 * 31.5f, 0.f), 63.f);
  float iy = fminf(fmaxf((float)y + a1 * 3.5f, 0.f), 7.f);
  float ix0 = floorf(ix), iy0 = floorf(iy);
  samp[idx * 4 + 0] = ix0;
  samp[idx * 4 + 1] = iy0;
  samp[idx * 4 + 2] = ix - ix0;
  samp[idx * 4 + 3] = iy - iy0;
}

// ---------------- bilinear warp of neighbors into FULL ---------------------
__global__ __launch_bounds__(256) void k_warp(const bf16_t* __restrict__ xbf,
                                              const float* __restrict__ samp,
                                              bf16_t* __restrict__ full) {
  int idx = blockIdx.x * 256 + threadIdx.x;        // 256*512
  int p = idx >> 9, n = idx & 511;
  int b = p >> 2, j = p & 3;
  int t = (j < 2) ? j : j + 1;                     // frame index == full slot
  float ix0f = samp[idx * 4 + 0], iy0f = samp[idx * 4 + 1];
  float wx = samp[idx * 4 + 2], wy = samp[idx * 4 + 3];
  int ix0 = (int)ix0f, iy0 = (int)iy0f;
  int ix1 = (ix0 + 1 > 63) ? 63 : ix0 + 1;
  int iy1 = (iy0 + 1 > 7) ? 7 : iy0 + 1;
  float w00 = (1.f - wx) * (1.f - wy), w01 = wx * (1.f - wy);
  float w10 = (1.f - wx) * wy, w11 = wx * wy;
  const bf16_t* src = xbf + (size_t)(b * F_ + t) * C_ * PX_;
  bf16_t* dst = full + ((size_t)b * KF_ + t * C_) * PX_ + n;
  int o00 = iy0 * W_ + ix0, o01 = iy0 * W_ + ix1;
  int o10 = iy1 * W_ + ix0, o11 = iy1 * W_ + ix1;
  for (int ch = 0; ch < C_; ++ch) {
    const bf16_t* s = src + (size_t)ch * PX_;
    float v = w00 * (float)s[o00] + w01 * (float)s[o01] +
              w10 * (float)s[o10] + w11 * (float)s[o11];
    dst[(size_t)ch * PX_] = (bf16_t)v;
  }
}

// ---------------- fusion 1x1 GEMM + BN + ReLU, M=256 N=512 K=1280 ----------
__global__ __launch_bounds__(256) void k_fusion(const bf16_t* __restrict__ full,
                                                const bf16_t* __restrict__ fw,
                                                const float* __restrict__ fb,
                                                const float* __restrict__ gamma,
                                                const float* __restrict__ beta,
                                                const float* __restrict__ mean,
                                                const float* __restrict__ var,
                                                bf16_t* __restrict__ fusedT) {
  __shared__ bf16_t As[128 * LDT];
  __shared__ bf16_t Bs[128 * LDT];
  const int tid = threadIdx.x, lane = tid & 31, wave = tid >> 5;
  const int wm = wave >> 1, wn = wave & 1;
  const int batch = blockIdx.z;
  const int mBase = blockIdx.y * 128;
  const int nBase = blockIdx.x * 128;
  const bf16_t* Bg = full + (size_t)batch * KF_ * PX_;
  const bf16_t* Ag = fw + (size_t)mBase * KF_;

  v8f acc[2][4];
#pragma unroll
  for (int i = 0; i < 2; ++i)
#pragma unroll
    for (int jj = 0; jj < 4; ++jj) acc[i][jj] = vzero8();

  for (int k0 = 0; k0 < KF_; k0 += 32) {
    stage_a_async(As, Ag, KF_, k0, tid);
    stage_b_vec(Bs, Bg, PX_, k0, nBase, tid);
    wait_async0();
    __syncthreads();
    v16bf af[2], bf[4];
#pragma unroll
    for (int i = 0; i < 2; ++i) af[i] = frag_a(As, wm * 32 + i * 16, lane);
#pragma unroll
    for (int jj = 0; jj < 4; ++jj) bf[jj] = frag_b(Bs, wn * 64 + jj * 16, lane);
#pragma unroll
    for (int i = 0; i < 2; ++i)
#pragma unroll
      for (int jj = 0; jj < 4; ++jj) acc[i][jj] = wmma_bf16(af[i], bf[jj], acc[i][jj]);
    __syncthreads();
  }
  // epilogue: bias + BN + relu ; scatter into linear-A layout [b][x][co*8+y]
  const int g = (lane >> 4) & 1, cl = lane & 15;
#pragma unroll
  for (int i = 0; i < 2; ++i)
#pragma unroll
    for (int jj = 0; jj < 4; ++jj) {
      int n = nBase + wn * 64 + jj * 16 + cl;
      int y = n >> 6, x = n & 63;
#pragma unroll
      for (int r = 0; r < 8; ++r) {
        int m = mBase + wm * 32 + i * 16 + g * 8 + r;
        float sc = gamma[m] * rsqrtf(var[m] + 1e-5f);
        float sh = beta[m] - mean[m] * sc;
        float v = acc[i][jj][r] + fb[m];
        v = fmaxf(v * sc + sh, 0.f);
        fusedT[((size_t)batch * W_ + x) * KL_ + m * H_ + y] = (bf16_t)v;
      }
    }
}

// ---------------- linear GEMM: M=4096 (b*w), N=256, K=2048 -----------------
__global__ __launch_bounds__(256) void k_linear(const bf16_t* __restrict__ fusedT,
                                                const bf16_t* __restrict__ lwT,
                                                const float* __restrict__ lb,
                                                float* __restrict__ out) {
  __shared__ bf16_t As[128 * LDT];
  __shared__ bf16_t Bs[128 * LDT];
  const int tid = threadIdx.x, lane = tid & 31, wave = tid >> 5;
  const int wm = wave >> 1, wn = wave & 1;
  const int mBase = blockIdx.y * 128;
  const int nBase = blockIdx.x * 128;
  const bf16_t* Ag = fusedT + (size_t)mBase * KL_;

  v8f acc[2][4];
#pragma unroll
  for (int i = 0; i < 2; ++i)
#pragma unroll
    for (int jj = 0; jj < 4; ++jj) acc[i][jj] = vzero8();

  for (int k0 = 0; k0 < KL_; k0 += 32) {
    stage_a_async(As, Ag, KL_, k0, tid);
    stage_b_vec(Bs, lwT, C_, k0, nBase, tid);
    wait_async0();
    __syncthreads();
    v16bf af[2], bf[4];
#pragma unroll
    for (int i = 0; i < 2; ++i) af[i] = frag_a(As, wm * 32 + i * 16, lane);
#pragma unroll
    for (int jj = 0; jj < 4; ++jj) bf[jj] = frag_b(Bs, wn * 64 + jj * 16, lane);
#pragma unroll
    for (int i = 0; i < 2; ++i)
#pragma unroll
      for (int jj = 0; jj < 4; ++jj) acc[i][jj] = wmma_bf16(af[i], bf[jj], acc[i][jj]);
    __syncthreads();
  }
  // epilogue: + bias ; out[b][co][0][x]
  const int g = (lane >> 4) & 1, cl = lane & 15;
#pragma unroll
  for (int i = 0; i < 2; ++i)
#pragma unroll
    for (int jj = 0; jj < 4; ++jj) {
      int co = nBase + wn * 64 + jj * 16 + cl;
#pragma unroll
      for (int r = 0; r < 8; ++r) {
        int m = mBase + wm * 32 + i * 16 + g * 8 + r;
        int b = m >> 6, x = m & 63;
        out[((size_t)b * C_ + co) * W_ + x] = acc[i][jj][r] + lb[co];
      }
    }
}

// ---------------------------------------------------------------------------
extern "C" void kernel_launch(void* const* d_in, const int* in_sizes, int n_in,
                              void* d_out, int out_size, void* d_ws, size_t ws_size,
                              hipStream_t stream) {
  const float* x        = (const float*)d_in[0];
  const float* flow_w1  = (const float*)d_in[1];
  const float* flow_b1  = (const float*)d_in[2];
  const float* flow_w2  = (const float*)d_in[3];
  const float* flow_b2  = (const float*)d_in[4];
  const float* fusion_w = (const float*)d_in[5];
  const float* fusion_b = (const float*)d_in[6];
  const float* bn_gamma = (const float*)d_in[7];
  const float* bn_beta  = (const float*)d_in[8];
  const float* bn_mean  = (const float*)d_in[9];
  const float* bn_var   = (const float*)d_in[10];
  const float* lin_w    = (const float*)d_in[11];
  const float* lin_b    = (const float*)d_in[12];
  float* out = (float*)d_out;

  bf16_t* XBF    = WSP(d_ws, OFF_XBF, bf16_t);
  bf16_t* A1     = WSP(d_ws, OFF_A1, bf16_t);
  bf16_t* HID    = WSP(d_ws, OFF_HID, bf16_t);
  float*  SAMP   = WSP(d_ws, OFF_SAMP, float);
  bf16_t* FULL   = WSP(d_ws, OFF_FULL, bf16_t);
  bf16_t* FUSW   = WSP(d_ws, OFF_FUSW, bf16_t);
  bf16_t* FUSEDT = WSP(d_ws, OFF_FUSEDT, bf16_t);
  bf16_t* LINWT  = WSP(d_ws, OFF_LINWT, bf16_t);

  // conversions / layout
  const int nx = (B_ * F_) * C_ * PX_;                         // 41,943,040
  k_cvt_bf16<<<nx / 256, 256, 0, stream>>>(x, XBF, nx);
  k_cvt_w1<<<(128 * K1_) / 256, 256, 0, stream>>>(flow_w1, A1);
  k_cvt_bf16<<<(C_ * KF_) / 256, 256, 0, stream>>>(fusion_w, FUSW, C_ * KF_);
  k_cvt_linw<<<(KL_ * C_) / 256, 256, 0, stream>>>(lin_w, LINWT);

  // flow estimation conv1 (WMMA im2col GEMM)
  k_conv1<<<dim3(4, PAIRS), 256, 0, stream>>>(XBF, A1, flow_b1, HID);
  // conv2 + sampling params
  k_flow2<<<(PAIRS * PX_) / 256, 256, 0, stream>>>(HID, flow_w2, flow_b2, SAMP);
  // warp neighbors + center copy into FULL
  k_warp<<<(PAIRS * PX_) / 256, 256, 0, stream>>>(XBF, SAMP, FULL);
  k_center<<<(B_ * C_ * PX_) / 256, 256, 0, stream>>>(XBF, FULL);
  // fusion 1x1 conv + BN + ReLU (WMMA)
  k_fusion<<<dim3(4, 2, B_), 256, 0, stream>>>(FULL, FUSW, fusion_b, bn_gamma,
                                               bn_beta, bn_mean, bn_var, FUSEDT);
  // final linear (WMMA) -> output in (b, c, 1, w)
  k_linear<<<dim3(2, 32), 256, 0, stream>>>(FUSEDT, LINWT, lin_b, out);
}
